// FractionalSelfAttention_69982197121578
// MI455X (gfx1250) — matrix operand order repaired and checked by reading, not compiled, not run on hardware
//
#include <hip/hip_runtime.h>
#include <hip/hip_bf16.h>
#include <math.h>

// Problem constants (from the reference)
constexpr int kB = 16;    // batch
constexpr int kS = 256;   // sequence
constexpr int kE = 4096;  // embed
constexpr int kH = 16;    // heads
constexpr int kD = 256;   // head dim (== kS)

typedef __attribute__((ext_vector_type(16))) __bf16 v16bf;
typedef __attribute__((ext_vector_type(8)))  float  v8f;
typedef __attribute__((ext_vector_type(4)))  float  v4f;

// ---------------------------------------------------------------------------
// WMMA fragment helpers per CDNA5 ISA 7.12.2 (wave32, v_wmma_f32_16x16x32_bf16)
//
// A 16x32 (MxK) bf16:  lane<16: row=lane, elems 0-15 -> K=0..15 (contiguous k, same order as B)
//                      lane>=16: row=lane-16, elems 0-15 -> K=16..31
// B 32x16 (KxN) bf16:  col = lane&15, elem e -> K = e + 16*(lane>>4)
// C/D 16x16 f32:       VGPR r -> row = r + 8*(lane>>4), col = lane&15
// ---------------------------------------------------------------------------

__device__ __forceinline__ int lane_id() { return threadIdx.x & 31; }

// A fragment from row-major bf16 (row stride ld elems). Per lane: one 32B run.
__device__ __forceinline__ v16bf load_a_bf16(const __bf16* src, int ld) {
    const int lane = lane_id();
    const int half = lane >> 4;
    const int row  = lane & 15;
    const __bf16* p = src + (size_t)row * ld;
    v16bf a;
#pragma unroll
    for (int e = 0; e < 16; ++e) {
        int kk = e + half * 16;
        a[e] = p[kk];
    }
    return a;
}

// B fragment where storage is B[n*ld + k] ("W^T view"): 32 contiguous bytes/lane.
__device__ __forceinline__ v16bf load_bT_bf16(const __bf16* src, int ld) {
    const int lane = lane_id();
    const int half = lane >> 4;
    const int col  = lane & 15;
    const __bf16* p = src + (size_t)col * ld + half * 16;
    v16bf b;
#pragma unroll
    for (int e = 0; e < 16; ++e) b[e] = p[e];
    return b;
}

__device__ __forceinline__ void store_c_f32(float* dst, int ld, v8f c) {
    const int lane = lane_id();
    const int half = lane >> 4;
    const int col  = lane & 15;
#pragma unroll
    for (int r = 0; r < 8; ++r)
        dst[(size_t)(r + half * 8) * ld + col] = c[r];
}

__device__ __forceinline__ void store_c_bf16(__bf16* dst, int ld, v8f c) {
    const int lane = lane_id();
    const int half = lane >> 4;
    const int col  = lane & 15;
#pragma unroll
    for (int r = 0; r < 8; ++r)
        dst[(size_t)(r + half * 8) * ld + col] = (__bf16)c[r];
}

// reuse-A hint must be an immediate -> template parameter
template <bool RA>
__device__ __forceinline__ v8f wmma_bf16(v16bf a, v16bf b, v8f c) {
    return __builtin_amdgcn_wmma_f32_16x16x32_bf16(
        /*neg_a=*/false, a, /*neg_b=*/false, b,
        /*c_mod=*/(short)0, c, /*reuse_a=*/RA, /*reuse_b=*/false);
}

// 4-wide N sweep sharing one A fragment: first 3 WMMAs hint A-reuse.
__device__ __forceinline__ void wmma_row4(v16bf a, const v16bf* bf, v8f* accRow) {
    accRow[0] = wmma_bf16<true >(a, bf[0], accRow[0]);
    accRow[1] = wmma_bf16<true >(a, bf[1], accRow[1]);
    accRow[2] = wmma_bf16<true >(a, bf[2], accRow[2]);
    accRow[3] = wmma_bf16<false>(a, bf[3], accRow[3]);
}

// ---------------------------------------------------------------------------
// Kernel 0: W f32 -> bf16 (row-major [n][k]; rows of W are columns of W^T)
// ---------------------------------------------------------------------------
__global__ void cvt_w_kernel(const float* __restrict__ w, __bf16* __restrict__ wb) {
    int i = (blockIdx.x * blockDim.x + threadIdx.x) * 4;
    v4f f = *(const v4f*)(w + i);
    wb[i + 0] = (__bf16)f.x;
    wb[i + 1] = (__bf16)f.y;
    wb[i + 2] = (__bf16)f.z;
    wb[i + 3] = (__bf16)f.w;
}

// ---------------------------------------------------------------------------
// Kernel 1: attention.
//   grid = 256 (x,y) pairs * 2 row-chunks of 128, block = 256 (8 waves, 2x4)
//   phase 1: scores = (Q*scale) @ K   (LDS-staged tiled WMMA GEMM)
//   phase 2: row softmax (f32 in LDS, in-place -> bf16 P)
//   phase 3: attn = P @ V             (LDS-staged tiled WMMA GEMM)
// Dynamic LDS: 128KB scores + 8KB Q-stage + 16KB B-stage = 152KB.
// ---------------------------------------------------------------------------
__global__ __launch_bounds__(256) void attn_kernel(const float* __restrict__ query,
                                                   const float* __restrict__ key,
                                                   const float* __restrict__ value,
                                                   __bf16* __restrict__ attn) {
    extern __shared__ char smem[];
    float*  s_scores = (float*)smem;                       // [128][256] f32 (later bf16 P in place)
    __bf16* s_a      = (__bf16*)(smem + 131072);           // [128][32]
    __bf16* s_b      = (__bf16*)(smem + 131072 + 8192);    // [256][32]  (B stored [n][k])

    const int pair  = blockIdx.x >> 1;
    const int chunk = blockIdx.x & 1;                      // 128-row chunk
    const int x = pair & 15;       // head index of aw == batch index of output rows
    const int y = pair >> 4;       // batch index of q,k == head index of v
    const int tid  = threadIdx.x;
    const int wave = tid >> 5;
    const int wm = wave >> 2;      // 0..1  (64 rows each)
    const int wn = wave & 3;       // 0..3  (64 cols each)
    const float scale = 0.0625f;   // D^-0.5

    const float* Qbase = query + (size_t)y * kS * kE + (size_t)x * kD
                               + (size_t)(chunk * 128) * kE;           // [i][t], ld kE
    const float* Kbase = key   + (size_t)y * kS * kE + (size_t)x * kD; // [t][j], ld kE
    const float* Vbase = value + (size_t)x * kS * kE + (size_t)y * kD; // [t][j], ld kE

    // ---------------- phase 1: scores = Qs @ K ----------------
    v8f acc[4][4];
#pragma unroll
    for (int mi = 0; mi < 4; ++mi)
#pragma unroll
        for (int ni = 0; ni < 4; ++ni) acc[mi][ni] = (v8f){};

#pragma unroll 1
    for (int k0 = 0; k0 < kD; k0 += 32) {
        // stage Q chunk 128x32 (scale folded), coalesced 16 f32 per thread
        {
            const int row = tid >> 1, hh = (tid & 1) * 16;
            const float* src = Qbase + (size_t)row * kE + k0 + hh;
            __bf16* dst = s_a + row * 32 + hh;
#pragma unroll
            for (int e = 0; e < 16; ++e) dst[e] = (__bf16)(src[e] * scale);
        }
        // stage K chunk 32x256 transposed into [n][k]
#pragma unroll 1
        for (int i = tid; i < 32 * 256; i += 256) {
            const int kk = i >> 8, n = i & 255;
            s_b[n * 32 + kk] = (__bf16)Kbase[(size_t)(k0 + kk) * kE + n];
        }
        __syncthreads();

        v16bf af[4], bf[4];
#pragma unroll
        for (int mi = 0; mi < 4; ++mi) af[mi] = load_a_bf16(s_a + (wm * 64 + mi * 16) * 32, 32);
#pragma unroll
        for (int ni = 0; ni < 4; ++ni) bf[ni] = load_bT_bf16(s_b + (wn * 64 + ni * 16) * 32, 32);
#pragma unroll
        for (int mi = 0; mi < 4; ++mi) wmma_row4(af[mi], bf, acc[mi]);
        __syncthreads();
    }
#pragma unroll
    for (int mi = 0; mi < 4; ++mi)
#pragma unroll
        for (int ni = 0; ni < 4; ++ni)
            store_c_f32(s_scores + (wm * 64 + mi * 16) * 256 + wn * 64 + ni * 16, 256, acc[mi][ni]);
    __syncthreads();

    // ---------------- phase 2: softmax per row (threads 0..127) ----------------
    if (tid < 128) {
        float* row = s_scores + tid * 256;
        float mx = -INFINITY;
#pragma unroll 4
        for (int j = 0; j < 256; ++j) mx = fmaxf(mx, row[j]);
        float sum = 0.f;
#pragma unroll 4
        for (int j = 0; j < 256; ++j) {
            float e = __expf(row[j] - mx);
            sum += e;
            row[j] = e;
        }
        const float inv = 1.f / sum;
        __bf16* prow = (__bf16*)row;   // in-place: write j only touches f32 j/2 (already read)
        for (int j = 0; j < 256; ++j) prow[j] = (__bf16)(row[j] * inv);
    }
    __syncthreads();

    // ---------------- phase 3: out = P @ V ----------------
#pragma unroll
    for (int mi = 0; mi < 4; ++mi)
#pragma unroll
        for (int ni = 0; ni < 4; ++ni) acc[mi][ni] = (v8f){};

#pragma unroll 1
    for (int k0 = 0; k0 < kD; k0 += 32) {
        // stage V chunk 32x256 transposed into [n][k]
#pragma unroll 1
        for (int i = tid; i < 32 * 256; i += 256) {
            const int kk = i >> 8, n = i & 255;
            s_b[n * 32 + kk] = (__bf16)Vbase[(size_t)(k0 + kk) * kE + n];
        }
        __syncthreads();

        v16bf af[4], bf[4];
#pragma unroll
        for (int mi = 0; mi < 4; ++mi)
            af[mi] = load_a_bf16((const __bf16*)(s_scores + (wm * 64 + mi * 16) * 256) + k0, 512);
#pragma unroll
        for (int ni = 0; ni < 4; ++ni) bf[ni] = load_bT_bf16(s_b + (wn * 64 + ni * 16) * 32, 32);
#pragma unroll
        for (int mi = 0; mi < 4; ++mi) wmma_row4(af[mi], bf, acc[mi]);
        __syncthreads();
    }

    // store bf16 to attn workspace at [x*256 + i][y*256 + j]
    __bf16* Obase = attn + (size_t)(x * kS + chunk * 128) * kE + (size_t)y * kD;
#pragma unroll
    for (int mi = 0; mi < 4; ++mi)
#pragma unroll
        for (int ni = 0; ni < 4; ++ni)
            store_c_bf16(Obase + (size_t)(wm * 64 + mi * 16) * kE + wn * 64 + ni * 16, kE, acc[mi][ni]);
}

// ---------------------------------------------------------------------------
// Kernel 2: out = attn(bf16) @ W^T + b  (4096x4096x4096, f32 out)
// grid 16x16 tiles of 256x256; block = 512 (16 waves as 4x4, 64x64 per wave).
// Per 32-K chunk: stage A 256x32 and B 32x256 (as [n][k]) in 32KB LDS.
// ---------------------------------------------------------------------------
__global__ __launch_bounds__(512) void out_proj_kernel(const __bf16* __restrict__ attn,
                                                       const __bf16* __restrict__ wb, // W bf16 [n][k]
                                                       const float* __restrict__ bias,
                                                       float* __restrict__ out) {
    __shared__ __bf16 s_a[256 * 32];   // 16KB, [m][k]
    __shared__ __bf16 s_b[256 * 32];   // 16KB, [n][k]

    const int tile_n = blockIdx.x;     // 0..15
    const int tile_m = blockIdx.y;     // 0..15
    const int tid  = threadIdx.x;
    const int wave = tid >> 5;
    const int wm = wave >> 2;          // 0..3
    const int wn = wave & 3;           // 0..3

    const __bf16* Abase = attn + (size_t)(tile_m * 256) * kE;
    const __bf16* Bbase = wb   + (size_t)(tile_n * 256) * kE;

    v8f acc[4][4];
#pragma unroll
    for (int mi = 0; mi < 4; ++mi)
#pragma unroll
        for (int ni = 0; ni < 4; ++ni) acc[mi][ni] = (v8f){};

#pragma unroll 1
    for (int k0 = 0; k0 < kE; k0 += 32) {
        // stage: 512 threads, each copies one aligned 32B chunk of A and of B
        {
            const int row = tid >> 1, hh = (tid & 1) * 16;
            *(v16bf*)(s_a + row * 32 + hh) = *(const v16bf*)(Abase + (size_t)row * kE + k0 + hh);
            *(v16bf*)(s_b + row * 32 + hh) = *(const v16bf*)(Bbase + (size_t)row * kE + k0 + hh);
        }
        __syncthreads();

        v16bf af[4], bf[4];
#pragma unroll
        for (int mi = 0; mi < 4; ++mi) af[mi] = load_a_bf16(s_a + (wm * 64 + mi * 16) * 32, 32);
#pragma unroll
        for (int ni = 0; ni < 4; ++ni) bf[ni] = load_bT_bf16(s_b + (wn * 64 + ni * 16) * 32, 32);
#pragma unroll
        for (int mi = 0; mi < 4; ++mi) wmma_row4(af[mi], bf, acc[mi]);
        __syncthreads();
    }

    // epilogue: + bias, f32 store
    const int lane = lane_id();
    const int half = lane >> 4;
    const int col  = lane & 15;
#pragma unroll
    for (int ni = 0; ni < 4; ++ni) {
        const int n = tile_n * 256 + wn * 64 + ni * 16 + col;
        const float bb = bias[n];
#pragma unroll
        for (int mi = 0; mi < 4; ++mi) {
            float* dst = out + (size_t)(tile_m * 256 + wm * 64 + mi * 16) * kE + n;
#pragma unroll
            for (int r = 0; r < 8; ++r)
                dst[(size_t)(r + half * 8) * kE] = acc[mi][ni][r] + bb;
        }
    }
}

// ---------------------------------------------------------------------------
extern "C" void kernel_launch(void* const* d_in, const int* in_sizes, int n_in,
                              void* d_out, int out_size, void* d_ws, size_t ws_size,
                              hipStream_t stream) {
    const float* query = (const float*)d_in[0];
    const float* key   = (const float*)d_in[1];
    const float* value = (const float*)d_in[2];
    const float* out_w = (const float*)d_in[3];
    const float* out_b = (const float*)d_in[4];
    float* out = (float*)d_out;

    // Workspace: [attn bf16: 16*256*4096] [Wb bf16: 4096*4096]
    __bf16* attn = (__bf16*)d_ws;
    __bf16* wb   = (__bf16*)((char*)d_ws + (size_t)kB * kS * kE * sizeof(__bf16));

    // 0) W -> bf16 (4 elems/thread)
    cvt_w_kernel<<<(kE * kE) / (256 * 4), 256, 0, stream>>>(out_w, wb);

    // 1) attention: 512 blocks, 152KB dynamic LDS (CDNA5 WGP has 320KB)
    constexpr int kAttnLds = 131072 + 8192 + 16384;
    static bool attr_set = [] {
        (void)hipFuncSetAttribute((const void*)attn_kernel,
                                  hipFuncAttributeMaxDynamicSharedMemorySize, kAttnLds);
        return true;
    }();
    (void)attr_set;
    attn_kernel<<<512, 256, kAttnLds, stream>>>(query, key, value, attn);

    // 2) out projection GEMM
    out_proj_kernel<<<dim3(16, 16), 512, 0, stream>>>(attn, wb, out_b, out);
}
